// MambaBlock_14637248545256
// MI455X (gfx1250) — matrix operand order repaired and checked by reading, not compiled, not run on hardware
//
#include <hip/hip_runtime.h>
#include <hip/hip_bf16.h>
#include <stdint.h>

#define BATCH   2
#define SEQ     1024
#define DMODEL  1024
#define DINNER  2048
#define DSTATE  16
#define DTRANK  64
#define DCONV   4
#define MROWS   (BATCH * SEQ)          // 2048 flattened rows
#define PROJW   (DTRANK + 2 * DSTATE)  // 96

typedef unsigned short ushort_t;
typedef __attribute__((ext_vector_type(16))) __bf16 v16bf;
typedef __attribute__((ext_vector_type(8)))  __bf16 v8bf;
typedef __attribute__((ext_vector_type(8)))  float  v8f;

__device__ __forceinline__ ushort_t f2bf(float f) {
  union { float f; uint32_t u; } v; v.f = f;
  uint32_t r = v.u + 0x7FFFu + ((v.u >> 16) & 1u);   // round-to-nearest-even
  return (ushort_t)(r >> 16);
}

// ---------------------------------------------------------------- RMSNorm -> bf16
__global__ void k_rmsnorm(const float* __restrict__ x, const float* __restrict__ w,
                          ushort_t* __restrict__ xn) {
  int row = blockIdx.x;
  const float* xr = x + (size_t)row * DMODEL;
  float ss = 0.f;
  for (int i = threadIdx.x; i < DMODEL; i += blockDim.x) { float v = xr[i]; ss += v * v; }
  for (int off = 16; off > 0; off >>= 1) ss += __shfl_down(ss, off, 32);
  __shared__ float red[8];
  __shared__ float scale_s;
  int wid = threadIdx.x >> 5, lane = threadIdx.x & 31;
  if (lane == 0) red[wid] = ss;
  __syncthreads();
  if (threadIdx.x == 0) {
    float t = 0.f;
    for (int i = 0; i < 8; ++i) t += red[i];
    scale_s = rsqrtf(t / (float)DMODEL + 1e-6f);
  }
  __syncthreads();
  float scale = scale_s;
  ushort_t* o = xn + (size_t)row * DMODEL;
  for (int i = threadIdx.x; i < DMODEL; i += blockDim.x)
    o[i] = f2bf(xr[i] * scale * w[i]);
}

// ---------------------------------------------------------------- fp32 -> bf16 convert
__global__ void k_f2bf(const float* __restrict__ s, ushort_t* __restrict__ d, int n) {
  int i = blockIdx.x * blockDim.x + threadIdx.x;
  if (i < n) d[i] = f2bf(s[i]);
}

// ---------------------------------------------------------------- big WMMA GEMM
// C[M,N] = A[M,K](bf16) * B[N,K]^T(bf16)  (+ addC), fp32 accumulate.
// Block = 8 waves (2 in M x 4 in N); each wave = 64x64 tile (4x4 WMMA frags).
// Block tile = 128 x 256. Requires: M % 128 == 0, N % 256 == 0, K % 32 == 0.
__global__ __launch_bounds__(256)
void k_gemm_bf16(const ushort_t* __restrict__ A, const ushort_t* __restrict__ B,
                 float* __restrict__ C, const float* __restrict__ addC,
                 int M, int N, int K) {
  const int lane  = threadIdx.x & 31;
  const int wave  = threadIdx.x >> 5;
  const int waveM = wave & 1;
  const int waveN = wave >> 1;
  const int m0 = blockIdx.y * 128 + waveM * 64;
  const int n0 = blockIdx.x * 256 + waveN * 64;
  const int half = lane >> 4;
  const int l15  = lane & 15;

  v8f zero = {};
  v8f acc[4][4];
#pragma unroll
  for (int i = 0; i < 4; ++i)
#pragma unroll
    for (int j = 0; j < 4; ++j) acc[i][j] = zero;

  const ushort_t* aBase[4];
  const ushort_t* bBase[4];
#pragma unroll
  for (int i = 0; i < 4; ++i) aBase[i] = A + (size_t)(m0 + i * 16 + l15) * K;
#pragma unroll
  for (int j = 0; j < 4; ++j) bBase[j] = B + (size_t)(n0 + j * 16 + l15) * K + 16 * half;

  for (int k0 = 0; k0 < K; k0 += 32) {
    if (k0 + 32 < K) {                       // near-cache prefetch of next K-chunk
      __builtin_prefetch(aBase[0] + k0 + 32, 0, 3);
      __builtin_prefetch(bBase[0] + k0 + 32, 0, 3);
    }
    union UA { v16bf v; v8bf h[2]; };
    v16bf a[4], b[4];
#pragma unroll
    for (int i = 0; i < 4; ++i) {            // A 16x32: 16 contiguous K at 16*half
      UA ua;
      ua.h[0] = *reinterpret_cast<const v8bf*>(aBase[i] + k0 + 16 * half);
      ua.h[1] = *reinterpret_cast<const v8bf*>(aBase[i] + k0 + 16 * half + 8);
      a[i] = ua.v;
    }
#pragma unroll
    for (int j = 0; j < 4; ++j)              // B 32x16: 16 contiguous K at 16*half
      b[j] = *reinterpret_cast<const v16bf*>(bBase[j] + k0);
#pragma unroll
    for (int i = 0; i < 4; ++i)
#pragma unroll
      for (int j = 0; j < 4; ++j)
        acc[i][j] = __builtin_amdgcn_wmma_f32_16x16x32_bf16(
            false, a[i], false, b[j], (short)0, acc[i][j], false, false);
  }

#pragma unroll
  for (int i = 0; i < 4; ++i)
#pragma unroll
    for (int j = 0; j < 4; ++j) {
      int col = n0 + j * 16 + l15;
#pragma unroll
      for (int r = 0; r < 8; ++r) {          // C row = r + 8*half
        int rowm = m0 + i * 16 + r + 8 * half;
        size_t idx = (size_t)rowm * N + col;
        float v = acc[i][j][r];
        if (addC) v += addC[idx];
        C[idx] = v;
      }
    }
}

// ---------------------------------------------------------------- narrow WMMA GEMM (N=96)
// One 16x16 tile per wave. Requires M,N % 16 == 0, K % 32 == 0.
__global__ __launch_bounds__(256)
void k_gemm_bf16_16(const ushort_t* __restrict__ A, const ushort_t* __restrict__ B,
                    float* __restrict__ C, int M, int N, int K) {
  int lane  = threadIdx.x & 31;
  int gwave = blockIdx.x * (blockDim.x >> 5) + (threadIdx.x >> 5);
  int tilesM = M >> 4, tilesN = N >> 4;
  if (gwave >= tilesM * tilesN) return;      // wave-uniform
  int tm = gwave % tilesM, tn = gwave / tilesM;
  int half = lane >> 4, l15 = lane & 15;
  const ushort_t* aBase = A + (size_t)(tm * 16 + l15) * K;
  const ushort_t* bBase = B + (size_t)(tn * 16 + l15) * K + 16 * half;
  v8f acc = {};
  for (int k0 = 0; k0 < K; k0 += 32) {
    union { v16bf v; v8bf h[2]; } ua;
    ua.h[0] = *reinterpret_cast<const v8bf*>(aBase + k0 + 16 * half);
    ua.h[1] = *reinterpret_cast<const v8bf*>(aBase + k0 + 16 * half + 8);
    v16bf b = *reinterpret_cast<const v16bf*>(bBase + k0);
    acc = __builtin_amdgcn_wmma_f32_16x16x32_bf16(
        false, ua.v, false, b, (short)0, acc, false, false);
  }
#pragma unroll
  for (int r = 0; r < 8; ++r)
    C[(size_t)(tm * 16 + r + 8 * half) * N + tn * 16 + l15] = acc[r];
}

// ---------------------------------------------------------------- causal depthwise conv + SiLU
__global__ void k_conv_silu(const float* __restrict__ xz, const float* __restrict__ cw,
                            const float* __restrict__ cb, float* __restrict__ u32,
                            ushort_t* __restrict__ ubf) {
  int idx = blockIdx.x * blockDim.x + threadIdx.x;
  if (idx >= BATCH * SEQ * DINNER) return;
  int c = idx & (DINNER - 1);
  int t = (idx >> 11) & (SEQ - 1);
  int b = idx >> 21;
  float acc = cb[c];
#pragma unroll
  for (int j = 0; j < DCONV; ++j) {
    int ts = t + j - (DCONV - 1);
    if (ts >= 0)
      acc += xz[(size_t)(b * SEQ + ts) * (2 * DINNER) + c] * cw[c * DCONV + j];
  }
  float s = acc / (1.f + __expf(-acc));      // silu
  u32[idx] = s;
  ubf[idx] = f2bf(s);
}

// ---------------------------------------------------------------- extract dt_raw slice -> bf16
__global__ void k_dtraw(const float* __restrict__ proj, ushort_t* __restrict__ out) {
  int i = blockIdx.x * blockDim.x + threadIdx.x;
  if (i >= MROWS * DTRANK) return;
  int row = i >> 6, c = i & (DTRANK - 1);
  out[i] = f2bf(proj[row * PROJW + c]);
}

// ---------------------------------------------------------------- softplus(dt_pre + bias)
__global__ void k_softplus(const float* __restrict__ pre, const float* __restrict__ bias,
                           float* __restrict__ dt) {
  int i = blockIdx.x * blockDim.x + threadIdx.x;
  if (i >= MROWS * DINNER) return;
  float v = pre[i] + bias[i & (DINNER - 1)];
  dt[i] = (v > 20.f) ? v : log1pf(__expf(v));
}

// ---------------------------------------------------------------- selective scan
// One thread per (batch, channel, state): 65536 threads. Dot over 16 states via
// wave-local 16-lane xor-shuffle reduction (wave32 -> 2 groups / wave).
__global__ void k_scan(const float* __restrict__ dt, const float* __restrict__ u,
                       const float* __restrict__ proj, const float* __restrict__ A_log,
                       const float* __restrict__ Dd, float* __restrict__ y) {
  int tid = blockIdx.x * blockDim.x + threadIdx.x;
  int n = tid & (DSTATE - 1);
  int d = (tid >> 4) & (DINNER - 1);
  int b = tid >> 15;
  float A  = -__expf(A_log[d * DSTATE + n]);
  float Dv = Dd[d];
  float s = 0.f;
  for (int t = 0; t < SEQ; ++t) {
    size_t row = (size_t)(b * SEQ + t);
    float dtv = dt[row * DINNER + d];
    float uv  = u [row * DINNER + d];
    float bv  = proj[row * PROJW + DTRANK + n];
    float cv  = proj[row * PROJW + DTRANK + DSTATE + n];
    float dA = __expf(dtv * A);
    s = dA * s + dtv * bv * uv;
    float p = s * cv;
    p += __shfl_xor(p, 1, 16);
    p += __shfl_xor(p, 2, 16);
    p += __shfl_xor(p, 4, 16);
    p += __shfl_xor(p, 8, 16);
    if (n == 0) y[row * DINNER + d] = p + Dv * uv;
  }
}

// ---------------------------------------------------------------- y * silu(z) -> bf16
__global__ void k_gate(const float* __restrict__ y, const float* __restrict__ xz,
                       ushort_t* __restrict__ ybf) {
  int i = blockIdx.x * blockDim.x + threadIdx.x;
  if (i >= MROWS * DINNER) return;
  int row = i >> 11, dcol = i & (DINNER - 1);
  float z = xz[(size_t)row * (2 * DINNER) + DINNER + dcol];
  float g = z / (1.f + __expf(-z));
  ybf[i] = f2bf(y[i] * g);
}

// ================================================================ launch
extern "C" void kernel_launch(void* const* d_in, const int* in_sizes, int n_in,
                              void* d_out, int out_size, void* d_ws, size_t ws_size,
                              hipStream_t stream) {
  const float* x         = (const float*)d_in[0];
  const float* norm_w    = (const float*)d_in[1];
  const float* in_proj_w = (const float*)d_in[2];
  const float* conv_w    = (const float*)d_in[3];
  const float* conv_b    = (const float*)d_in[4];
  const float* x_proj_w  = (const float*)d_in[5];
  const float* dt_proj_w = (const float*)d_in[6];
  const float* dt_proj_b = (const float*)d_in[7];
  const float* A_log     = (const float*)d_in[8];
  const float* D_diag    = (const float*)d_in[9];
  const float* out_proj_w= (const float*)d_in[10];
  float* out = (float*)d_out;

  char* ws = (char*)d_ws;
  size_t off = 0;
  auto alloc = [&](size_t bytes) -> void* {
    void* p = (void*)(ws + off);
    off += (bytes + 255) & ~(size_t)255;
    return p;
  };

  ushort_t* xn_bf    = (ushort_t*)alloc((size_t)MROWS * DMODEL * 2);
  ushort_t* w_in_bf  = (ushort_t*)alloc((size_t)2 * DINNER * DMODEL * 2);
  ushort_t* w_x_bf   = (ushort_t*)alloc((size_t)PROJW * DINNER * 2);
  ushort_t* w_dt_bf  = (ushort_t*)alloc((size_t)DINNER * DTRANK * 2);
  ushort_t* w_out_bf = (ushort_t*)alloc((size_t)DMODEL * DINNER * 2);
  float*    xz       = (float*)   alloc((size_t)MROWS * 2 * DINNER * 4);
  ushort_t* u_bf     = (ushort_t*)alloc((size_t)MROWS * DINNER * 2);
  float*    u_f32    = (float*)   alloc((size_t)MROWS * DINNER * 4);
  float*    proj     = (float*)   alloc((size_t)MROWS * PROJW * 4);
  ushort_t* dtraw_bf = (ushort_t*)alloc((size_t)MROWS * DTRANK * 2);
  float*    dt_pre   = (float*)   alloc((size_t)MROWS * DINNER * 4);
  float*    dt_sp    = (float*)   alloc((size_t)MROWS * DINNER * 4);
  float*    y_scan   = (float*)   alloc((size_t)MROWS * DINNER * 4);
  ushort_t* y_bf     = (ushort_t*)alloc((size_t)MROWS * DINNER * 2);

  // 1) RMSNorm -> bf16 activations
  k_rmsnorm<<<MROWS, 256, 0, stream>>>(x, norm_w, xn_bf);

  // 2) weights -> bf16
  int nw;
  nw = 2 * DINNER * DMODEL; k_f2bf<<<(nw + 255) / 256, 256, 0, stream>>>(in_proj_w, w_in_bf, nw);
  nw = PROJW * DINNER;      k_f2bf<<<(nw + 255) / 256, 256, 0, stream>>>(x_proj_w,  w_x_bf,  nw);
  nw = DINNER * DTRANK;     k_f2bf<<<(nw + 255) / 256, 256, 0, stream>>>(dt_proj_w, w_dt_bf, nw);
  nw = DMODEL * DINNER;     k_f2bf<<<(nw + 255) / 256, 256, 0, stream>>>(out_proj_w, w_out_bf, nw);

  // 3) in_proj: xz[2048 x 4096]
  dim3 g1((2 * DINNER) / 256, MROWS / 128);
  k_gemm_bf16<<<g1, 256, 0, stream>>>(xn_bf, w_in_bf, xz, nullptr, MROWS, 2 * DINNER, DMODEL);

  // 4) depthwise causal conv + SiLU
  int nconv = BATCH * SEQ * DINNER;
  k_conv_silu<<<(nconv + 255) / 256, 256, 0, stream>>>(xz, conv_w, conv_b, u_f32, u_bf);

  // 5) x_proj: proj[2048 x 96]
  int tiles = (MROWS / 16) * (PROJW / 16);
  k_gemm_bf16_16<<<(tiles + 7) / 8, 256, 0, stream>>>(u_bf, w_x_bf, proj, MROWS, PROJW, DINNER);

  // 6) dt_raw slice -> bf16, dt_proj GEMM, softplus
  int ndr = MROWS * DTRANK;
  k_dtraw<<<(ndr + 255) / 256, 256, 0, stream>>>(proj, dtraw_bf);
  dim3 g3(DINNER / 256, MROWS / 128);
  k_gemm_bf16<<<g3, 256, 0, stream>>>(dtraw_bf, w_dt_bf, dt_pre, nullptr, MROWS, DINNER, DTRANK);
  int nel = MROWS * DINNER;
  k_softplus<<<(nel + 255) / 256, 256, 0, stream>>>(dt_pre, dt_proj_b, dt_sp);

  // 7) selective scan
  k_scan<<<(BATCH * DINNER * DSTATE) / 256, 256, 0, stream>>>(dt_sp, u_f32, proj, A_log, D_diag, y_scan);

  // 8) gate with silu(z) -> bf16
  k_gate<<<(nel + 255) / 256, 256, 0, stream>>>(y_scan, xz, y_bf);

  // 9) out_proj + residual (fused into GEMM epilogue), writes d_out
  dim3 g4(DMODEL / 256, MROWS / 128);
  k_gemm_bf16<<<g4, 256, 0, stream>>>(y_bf, w_out_bf, out, x, MROWS, DMODEL, DINNER);
}
